// FineGrainedRetriever_27668179321262
// MI455X (gfx1250) — matrix operand order repaired and checked by reading, not compiled, not run on hardware
//
#include <hip/hip_runtime.h>

// ---------------------------------------------------------------------------
// FineGrainedRetriever for MI455X (gfx1250, wave32, WMMA)
// bf16 WMMA GEMMs with f32 accumulation; edge-level matmuls factored into
// node-level / relation-level GEMMs + gathers.
// ---------------------------------------------------------------------------

typedef __attribute__((ext_vector_type(16))) __bf16 v16bf;
typedef __attribute__((ext_vector_type(8)))  __bf16 v8bf;
typedef __attribute__((ext_vector_type(4)))  __bf16 v4bf;
typedef __attribute__((ext_vector_type(8)))  float  v8f;
typedef __attribute__((ext_vector_type(4)))  float  v4f;

#define EMBD 1024

__device__ __forceinline__ void atomAddF(float* p, float v) {
  __hip_atomic_fetch_add(p, v, __ATOMIC_RELAXED, __HIP_MEMORY_SCOPE_AGENT);
}

__device__ __forceinline__ unsigned fgr_hash32(unsigned x) {
  x = x * 747796405u + 2891336453u;
  x = ((x >> ((x >> 28) + 4u)) ^ x) * 277803737u;
  return (x >> 22) ^ x;
}

// ---------------------------------------------------------------------------
// GEMM: C[M,N] = A[M,K](lda) @ B[K,N](ldb)  (+bias, optional relu)
// Block: 128 threads (4 waves), tile 64(M) x 128(N), K-step 32.
// Waves arranged 2x2; each wave computes 32x64 via 2x4 = 8 WMMAs per K-step.
// LDS: A row-major (96B stride), B TRANSPOSED col-major (96B stride) so every
// fragment is a contiguous, 16B/32B-aligned ds_load_b128 — no v_mov_b16 packs.
// ---------------------------------------------------------------------------
#define TILE_M 64
#define TILE_N 128
#define TILE_K 32
#define LSTR   48   // __bf16 elements per LDS row = 96 bytes

__global__ __launch_bounds__(128) void fgr_gemm_wmma(
    const float* __restrict__ A, int lda,
    const float* __restrict__ B, int ldb,
    float* __restrict__ C, int ldc,
    int M, int N, int Kd,
    const float* __restrict__ bias, int relu)
{
  __shared__ __bf16 sA[TILE_M][LSTR];    // 64 rows  x 32 K  (6 KB)
  __shared__ __bf16 sBt[TILE_N][LSTR];   // 128 cols x 32 K  (12 KB, transposed)

  const int tid   = threadIdx.x;         // 0..127
  const int wave  = tid >> 5;            // 0..3
  const int lane  = tid & 31;            // wave32
  const int l16   = lane & 15;
  const int hi16  = lane >> 4;           // 0 | 1
  const int tileM = blockIdx.y * TILE_M;
  const int tileN = blockIdx.x * TILE_N;
  const int wm    = (wave >> 1) * 32;    // wave row offset (0|32)
  const int wn    = (wave & 1) * 64;     // wave col offset (0|64)

  v8f acc[2][4] = {};

  for (int k0 = 0; k0 < Kd; k0 += TILE_K) {
    if (k0 + TILE_K < Kd) {   // speculative prefetch of next K-tile
      int pr = tileM + (tid & 63); if (pr >= M) pr = M - 1;
      __builtin_prefetch(A + (size_t)pr * lda + k0 + TILE_K, 0, 1);
      __builtin_prefetch(B + (size_t)(k0 + TILE_K + (tid & 31)) * ldb + tileN, 0, 1);
    }

    // ---- stage A: 64x32 floats, vector loads, packed v4bf stores ----
    for (int i = tid; i < (TILE_M * TILE_K) / 4; i += 128) {  // 512 quads
      int r  = i >> 3;            // / (32/4)
      int c4 = (i & 7) * 4;
      int gr = tileM + r, gc = k0 + c4;
      v4f v = {0.f, 0.f, 0.f, 0.f};
      if (gr < M && gc + 3 < Kd)
        v = *(const v4f*)(A + (size_t)gr * lda + gc);
      v4bf pk;
      pk[0] = (__bf16)v[0]; pk[1] = (__bf16)v[1];
      pk[2] = (__bf16)v[2]; pk[3] = (__bf16)v[3];
      *(v4bf*)&sA[r][c4] = pk;    // 8B-aligned ds_store_b64
    }
    // ---- stage B transposed: lanes sweep columns (coalesced), 4 K each ----
    for (int i = tid; i < 8 * TILE_N; i += 128) {             // 1024 quads
      int col = i & (TILE_N - 1);
      int kg  = (i >> 7) * 4;
      int gc  = tileN + col;
      v4bf pk;
      #pragma unroll
      for (int u = 0; u < 4; ++u) {
        int gk = k0 + kg + u;
        float v = (gk < Kd && gc < N) ? B[(size_t)gk * ldb + gc] : 0.f;
        pk[u] = (__bf16)v;
      }
      *(v4bf*)&sBt[col][kg] = pk; // 8B-aligned ds_store_b64
    }
    __syncthreads();

    // ---- fragments (this kernel's contiguous k order, A and B alike) ----
    // A 16x32: lanes 0-15 hold K 0..15; lanes 16-31 K 16..31
    // B 32x16: lanes 0-15 hold K 0..15; lanes 16-31 K 16..31
    const int kA = hi16 * 16;
    const int kB = hi16 * 16;
    v16bf afrag[2], bfrag[4];
    #pragma unroll
    for (int t = 0; t < 2; ++t) {
      int arow = wm + t * 16 + l16;
      v8bf alo = *(const v8bf*)&sA[arow][kA];        // 16B aligned b128
      v8bf ahi = *(const v8bf*)&sA[arow][8 + kA];    // 16B aligned b128
      afrag[t] = __builtin_shufflevector(alo, ahi,
          0, 1, 2, 3, 4, 5, 6, 7, 8, 9, 10, 11, 12, 13, 14, 15);
    }
    #pragma unroll
    for (int t = 0; t < 4; ++t) {
      int bcol = wn + t * 16 + l16;
      bfrag[t] = *(const v16bf*)&sBt[bcol][kB];      // 32B aligned, 2x b128
    }
    #pragma unroll
    for (int ti = 0; ti < 2; ++ti)
      #pragma unroll
      for (int tj = 0; tj < 4; ++tj)
        acc[ti][tj] = __builtin_amdgcn_wmma_f32_16x16x32_bf16(
            false, afrag[ti], false, bfrag[tj], (short)0, acc[ti][tj],
            false, false);
    __syncthreads();
  }

  // ---- epilogue: D layout — VGPR v holds row (v + 8*hi16), col = lane&15 ----
  #pragma unroll
  for (int ti = 0; ti < 2; ++ti)
    #pragma unroll
    for (int tj = 0; tj < 4; ++tj) {
      int colg = tileN + wn + tj * 16 + l16;
      if (colg >= N) continue;
      float bv = bias ? bias[colg] : 0.f;
      #pragma unroll
      for (int v = 0; v < 8; ++v) {
        int rowg = tileM + wm + ti * 16 + v + hi16 * 8;
        if (rowg < M) {
          float x = acc[ti][tj][v] + bv;
          if (relu) x = x > 0.f ? x : 0.f;
          C[(size_t)rowg * ldc + colg] = x;
        }
      }
    }
}

// ---------------------------------------------------------------------------
// Small helper kernels
// ---------------------------------------------------------------------------
__global__ void fgr_zero(float* p, int n) {
  int i = blockIdx.x * blockDim.x + threadIdx.x;
  if (i < n) p[i] = 0.f;
}

__global__ void fgr_build_he(const float* __restrict__ ent,
                             const float* __restrict__ nontext,
                             float* __restrict__ he, int n_text) {
  int i = blockIdx.x;
  const float* s = (i < n_text) ? ent + (size_t)i * EMBD : nontext;
  for (int j = threadIdx.x; j < EMBD; j += blockDim.x)
    he[(size_t)i * EMBD + j] = s[j];
}

__global__ void fgr_deg_count(const int* __restrict__ h_id,
                              const int* __restrict__ t_id, int E,
                              float* __restrict__ deg) {
  int e = blockIdx.x * blockDim.x + threadIdx.x;
  if (e >= 2 * E) return;
  int dst = (e < E) ? t_id[e] : h_id[e - E];
  atomAddF(&deg[dst], 1.0f);
}

// rank-2 update for the sparse topic one-hot columns
__global__ void fgr_add_topic(float* __restrict__ buf,
                              const float* __restrict__ topic,
                              const float* __restrict__ r0,
                              const float* __restrict__ r1) {
  int i = blockIdx.x;
  float t0 = topic[2 * i], t1 = topic[2 * i + 1];
  if (t0 == 0.f && t1 == 0.f) return;   // almost always
  for (int j = threadIdx.x; j < EMBD; j += blockDim.x)
    buf[(size_t)i * EMBD + j] += t0 * r0[j] + t1 * r1[j];
}

// per-edge message: relu(nodemsg[src] + relmsg[rel] + b) atomically summed to dst
__global__ __launch_bounds__(256) void fgr_msg_agg(
    const int* __restrict__ h_id, const int* __restrict__ t_id,
    const int* __restrict__ r_id, int E,
    const float* __restrict__ nodemsg,
    const float* __restrict__ relF, const float* __restrict__ relR,
    const float* __restrict__ bmsg, float* __restrict__ agg)
{
  int e = blockIdx.x;
  bool fwd = e < E;
  int eid = fwd ? e : e - E;
  int src = fwd ? h_id[eid] : t_id[eid];
  int dst = fwd ? t_id[eid] : h_id[eid];
  int rel = r_id[eid];
  const float* nm = nodemsg + (size_t)src * EMBD;
  const float* rm = (fwd ? relF : relR) + (size_t)rel * EMBD;
  float* ag = agg + (size_t)dst * EMBD;
  for (int j = threadIdx.x; j < EMBD; j += 256) {
    float m = nm[j] + rm[j] + bmsg[j];
    m = m > 0.f ? m : 0.f;
    atomAddF(&ag[j], m);    // hardware global_atomic_add_f32
  }
}

// h_out = relu(selfpre + b_self + agg / max(deg,1)); strided write into h_node
__global__ void fgr_combine(const float* __restrict__ selfpre,
                            const float* __restrict__ bself,
                            const float* __restrict__ agg,
                            const float* __restrict__ deg,
                            float* __restrict__ hout, int ldh) {
  int i = blockIdx.x;
  float d = deg[i]; if (d < 1.f) d = 1.f;
  float inv = 1.f / d;
  for (int j = threadIdx.x; j < EMBD; j += blockDim.x) {
    float x = selfpre[(size_t)i * EMBD + j] + bself[j]
            + agg[(size_t)i * EMBD + j] * inv;
    hout[(size_t)i * ldh + j] = x > 0.f ? x : 0.f;
  }
}

// qW[j] = sum_k q[k] * W[k,j]   (q block of pred_w1)
__global__ void fgr_qw(const float* __restrict__ q,
                       const float* __restrict__ W,
                       float* __restrict__ out) {
  int j = blockIdx.x * blockDim.x + threadIdx.x;
  if (j >= EMBD) return;
  float acc = 0.f;
  for (int k = 0; k < EMBD; ++k) acc += q[k] * W[(size_t)k * EMBD + j];
  out[j] = acc;
}

// per-edge: logit = relu(qW + srcP[src] + relP[rel] + dstP[dst] + b1) . w2 + b2
// y = logit + gumbel(hash(e)); store order-preserving uint key.
__global__ __launch_bounds__(256) void fgr_edge_logit(
    const int* __restrict__ h_id, const int* __restrict__ t_id,
    const int* __restrict__ r_id, int E,
    const float* __restrict__ qW,
    const float* __restrict__ srcP, const float* __restrict__ dstP,
    const float* __restrict__ relF, const float* __restrict__ relR,
    const float* __restrict__ b1, const float* __restrict__ w2,
    const float* __restrict__ b2,
    float* __restrict__ yv, unsigned* __restrict__ keys)
{
  __shared__ float red[256];
  int e = blockIdx.x;
  bool fwd = e < E;
  int eid = fwd ? e : e - E;
  int src = fwd ? h_id[eid] : t_id[eid];
  int dst = fwd ? t_id[eid] : h_id[eid];
  int rel = r_id[eid];
  const float* sp = srcP + (size_t)src * EMBD;
  const float* dp = dstP + (size_t)dst * EMBD;
  const float* rp = (fwd ? relF : relR) + (size_t)rel * EMBD;
  float acc = 0.f;
  for (int j = threadIdx.x; j < EMBD; j += 256) {
    float h = qW[j] + sp[j] + rp[j] + dp[j] + b1[j];
    h = h > 0.f ? h : 0.f;
    acc += h * w2[j];
  }
  red[threadIdx.x] = acc;
  __syncthreads();
  for (int s = 128; s > 0; s >>= 1) {
    if (threadIdx.x < s) red[threadIdx.x] += red[threadIdx.x + s];
    __syncthreads();
  }
  if (threadIdx.x == 0) {
    float logit = red[0] + b2[0];
    unsigned hh = fgr_hash32((unsigned)e ^ 0x9E3779B9u);
    float u = ((float)hh + 0.5f) * 2.3283064365386963e-10f;
    u = fminf(fmaxf(u, 1e-10f), 1.f - 1e-10f);
    float g = -logf(-logf(u));
    float y = logit + g;
    yv[e] = y;
    unsigned ub = __float_as_uint(y);
    keys[e] = (ub & 0x80000000u) ? ~ub : (ub | 0x80000000u);
  }
}

// single-workgroup 4-pass radix select of K-th largest key.
// sel[0]=threshold, sel[1]=countGreater, sel[2]=tie counter (reset), sel[3]=tiesNeeded
__global__ __launch_bounds__(1024) void fgr_topk_select(
    const unsigned* __restrict__ keys, int n, int K, unsigned* __restrict__ sel)
{
  __shared__ unsigned hist[256];
  __shared__ unsigned prefix;
  __shared__ int remK;
  if (threadIdx.x == 0) { prefix = 0; remK = K; }
  __syncthreads();
  for (int shift = 24; shift >= 0; shift -= 8) {
    if (threadIdx.x < 256) hist[threadIdx.x] = 0;
    __syncthreads();
    unsigned pmask = (shift == 24) ? 0u : (0xFFFFFFFFu << (shift + 8));
    unsigned pref = prefix;
    for (int i = threadIdx.x; i < n; i += blockDim.x) {
      unsigned k = keys[i];
      if ((k & pmask) == (pref & pmask))
        atomicAdd(&hist[(k >> shift) & 0xFFu], 1u);
    }
    __syncthreads();
    if (threadIdx.x == 0) {
      int r = remK;
      int b = 255;
      for (; b > 0; --b) {
        int c = (int)hist[b];
        if (c >= r) break;
        r -= c;
      }
      prefix |= ((unsigned)b) << shift;
      remK = r;
    }
    __syncthreads();
  }
  if (threadIdx.x == 0) {
    sel[0] = prefix;
    sel[1] = (unsigned)(K - remK);
    sel[2] = 0u;               // tie counter, re-zeroed every call
    sel[3] = (unsigned)remK;   // number of == threshold entries to take
  }
}

// output = hard one-hot (hard - sg(soft) + soft == hard numerically)
__global__ void fgr_scatter(const unsigned* __restrict__ keys, int n,
                            unsigned* __restrict__ sel,
                            float* __restrict__ out)
{
  int i = blockIdx.x * blockDim.x + threadIdx.x;
  if (i >= n) return;
  unsigned T = sel[0];
  unsigned k = keys[i];
  float v = 0.f;
  if (k > T) v = 1.f;
  else if (k == T) {
    unsigned pos = atomicAdd(&sel[2], 1u);
    if (pos < sel[3]) v = 1.f;
  }
  out[i] = v;
}

// ---------------------------------------------------------------------------
// Host orchestration
// ---------------------------------------------------------------------------
static void launch_gemm(hipStream_t s, const float* A, int lda,
                        const float* B, int ldb, float* C, int ldc,
                        int M, int N, int Kd, const float* bias, int relu) {
  dim3 g((N + TILE_N - 1) / TILE_N, (M + TILE_M - 1) / TILE_M);
  fgr_gemm_wmma<<<g, dim3(128), 0, s>>>(A, lda, B, ldb, C, ldc, M, N, Kd, bias, relu);
}

extern "C" void kernel_launch(void* const* d_in, const int* in_sizes, int n_in,
                              void* d_out, int out_size, void* d_ws, size_t ws_size,
                              hipStream_t stream) {
  (void)n_in; (void)out_size; (void)ws_size;
  const int* h_id = (const int*)d_in[0];
  const int* t_id = (const int*)d_in[1];
  const int* r_id = (const int*)d_in[2];
  // d_in[3] = num_non_text_entities (device scalar; derived from sizes instead)
  const float* q_emb      = (const float*)d_in[4];
  const float* entity     = (const float*)d_in[5];
  const float* relation   = (const float*)d_in[6];
  const float* topic      = (const float*)d_in[7];
  const float* non_text   = (const float*)d_in[8];
  const float* w_self1    = (const float*)d_in[9];
  const float* b_self1    = (const float*)d_in[10];
  const float* w_msg1     = (const float*)d_in[11];
  const float* b_msg1     = (const float*)d_in[12];
  const float* w_self2    = (const float*)d_in[13];
  const float* b_self2    = (const float*)d_in[14];
  const float* w_msg2     = (const float*)d_in[15];
  const float* b_msg2     = (const float*)d_in[16];
  const float* rev_w1     = (const float*)d_in[17];
  const float* rev_b1     = (const float*)d_in[18];
  const float* rev_w2     = (const float*)d_in[19];
  const float* rev_b2     = (const float*)d_in[20];
  const float* pred_w1    = (const float*)d_in[21];
  const float* pred_b1    = (const float*)d_in[22];
  const float* pred_w2    = (const float*)d_in[23];
  const float* pred_b2    = (const float*)d_in[24];

  const int E       = in_sizes[0];
  const int n_text  = in_sizes[5] / EMBD;
  const int n_nodes = in_sizes[7] / 2;
  const int n_rel   = in_sizes[6] / EMBD;
  const int TWO_E   = 2 * E;
  const int TOPK    = 512;

  // workspace carve-up (256B aligned)
  char* wsb = (char*)d_ws;
  size_t off = 0;
  auto carve = [&](size_t bytes) -> void* {
    void* p = wsb + off;
    off = (off + bytes + 255) & ~(size_t)255;
    return p;
  };
  const size_t SZ_NODE = (size_t)n_nodes * EMBD * sizeof(float);
  const size_t SZ_REL  = (size_t)n_rel  * EMBD * sizeof(float);
  float* B0 = (float*)carve(SZ_NODE);            // h_e, later src_pre
  float* B1 = (float*)carve(SZ_NODE);            // selfpre, later dst_pre
  float* B2 = (float*)carve(SZ_NODE);            // nodemsg
  float* B3 = (float*)carve(SZ_NODE);            // agg
  float* HN = (float*)carve(SZ_NODE * 2);        // h_node [n_nodes, 2048]
  float* R0 = (float*)carve(SZ_REL);             // rev_hidden
  float* R1 = (float*)carve(SZ_REL);             // rev_table
  float* R2 = (float*)carve(SZ_REL);             // rel scratch fwd (reused 3x)
  float* R3 = (float*)carve(SZ_REL);             // rel scratch rev (reused 3x)
  float* deg  = (float*)carve((size_t)n_nodes * sizeof(float));
  float* qW   = (float*)carve(EMBD * sizeof(float));
  float* yv   = (float*)carve((size_t)TWO_E * sizeof(float));
  unsigned* keys = (unsigned*)carve((size_t)TWO_E * sizeof(unsigned));
  unsigned* sel  = (unsigned*)carve(4 * sizeof(unsigned));

  const dim3 b256(256);
  // node features & degrees
  fgr_build_he<<<n_nodes, b256, 0, stream>>>(entity, non_text, B0, n_text);
  fgr_zero<<<(n_nodes + 255) / 256, b256, 0, stream>>>(deg, n_nodes);
  fgr_deg_count<<<(TWO_E + 255) / 256, b256, 0, stream>>>(h_id, t_id, E, deg);

  // reverse-relation MLP on the relation table (not per edge)
  launch_gemm(stream, relation, EMBD, rev_w1, EMBD, R0, EMBD, n_rel, EMBD, EMBD, rev_b1, 1);
  launch_gemm(stream, R0, EMBD, rev_w2, EMBD, R1, EMBD, n_rel, EMBD, EMBD, rev_b2, 0);

  // ---- SAGE layer 1 ----
  launch_gemm(stream, B0, EMBD, w_self1, EMBD, B1, EMBD, n_nodes, EMBD, EMBD, nullptr, 0);
  fgr_add_topic<<<n_nodes, b256, 0, stream>>>(B1, topic,
      w_self1 + (size_t)1024 * EMBD, w_self1 + (size_t)1025 * EMBD);
  launch_gemm(stream, B0, EMBD, w_msg1, EMBD, B2, EMBD, n_nodes, EMBD, EMBD, nullptr, 0);
  fgr_add_topic<<<n_nodes, b256, 0, stream>>>(B2, topic,
      w_msg1 + (size_t)1024 * EMBD, w_msg1 + (size_t)1025 * EMBD);
  launch_gemm(stream, relation, EMBD, w_msg1 + (size_t)1026 * EMBD, EMBD, R2, EMBD,
              n_rel, EMBD, EMBD, nullptr, 0);
  launch_gemm(stream, R1, EMBD, w_msg1 + (size_t)1026 * EMBD, EMBD, R3, EMBD,
              n_rel, EMBD, EMBD, nullptr, 0);
  fgr_zero<<<((int)((size_t)n_nodes * EMBD) + 255) / 256, b256, 0, stream>>>(B3, n_nodes * EMBD);
  fgr_msg_agg<<<TWO_E, b256, 0, stream>>>(h_id, t_id, r_id, E, B2, R2, R3, b_msg1, B3);
  fgr_combine<<<n_nodes, b256, 0, stream>>>(B1, b_self1, B3, deg, HN, 2048);

  // ---- SAGE layer 2 (h1 = first half of HN, lda = 2048) ----
  launch_gemm(stream, HN, 2048, w_self2, EMBD, B1, EMBD, n_nodes, EMBD, EMBD, nullptr, 0);
  launch_gemm(stream, HN, 2048, w_msg2, EMBD, B2, EMBD, n_nodes, EMBD, EMBD, nullptr, 0);
  launch_gemm(stream, relation, EMBD, w_msg2 + (size_t)1024 * EMBD, EMBD, R2, EMBD,
              n_rel, EMBD, EMBD, nullptr, 0);
  launch_gemm(stream, R1, EMBD, w_msg2 + (size_t)1024 * EMBD, EMBD, R3, EMBD,
              n_rel, EMBD, EMBD, nullptr, 0);
  fgr_zero<<<((int)((size_t)n_nodes * EMBD) + 255) / 256, b256, 0, stream>>>(B3, n_nodes * EMBD);
  fgr_msg_agg<<<TWO_E, b256, 0, stream>>>(h_id, t_id, r_id, E, B2, R2, R3, b_msg2, B3);
  fgr_combine<<<n_nodes, b256, 0, stream>>>(B1, b_self2, B3, deg, HN + 1024, 2048);

  // ---- predictor, factored: qW + hn@Wsrc (gather) + rel@Wr (gather) + hn@Wdst ----
  launch_gemm(stream, HN, 2048, pred_w1 + (size_t)1024 * EMBD, EMBD, B0, EMBD,
              n_nodes, EMBD, 2048, nullptr, 0);                       // src_pre
  launch_gemm(stream, HN, 2048, pred_w1 + (size_t)4096 * EMBD, EMBD, B1, EMBD,
              n_nodes, EMBD, 2048, nullptr, 0);                       // dst_pre
  launch_gemm(stream, relation, EMBD, pred_w1 + (size_t)3072 * EMBD, EMBD, R2, EMBD,
              n_rel, EMBD, EMBD, nullptr, 0);                         // relpred fwd
  launch_gemm(stream, R1, EMBD, pred_w1 + (size_t)3072 * EMBD, EMBD, R3, EMBD,
              n_rel, EMBD, EMBD, nullptr, 0);                         // relpred rev
  fgr_qw<<<(EMBD + 255) / 256, b256, 0, stream>>>(q_emb, pred_w1, qW);

  fgr_edge_logit<<<TWO_E, b256, 0, stream>>>(h_id, t_id, r_id, E, qW, B0, B1,
                                             R2, R3, pred_b1, pred_w2, pred_b2,
                                             yv, keys);
  fgr_topk_select<<<1, 1024, 0, stream>>>(keys, TWO_E, TOPK, sel);
  fgr_scatter<<<(TWO_E + 255) / 256, b256, 0, stream>>>(keys, TWO_E, sel, (float*)d_out);
}
